// ProteinFeaturesLigand_3925600108551
// MI455X (gfx1250) — matrix operand order repaired and checked by reading, not compiled, not run on hardware
//
#include <hip/hip_runtime.h>
#include <hip/hip_bf16.h>

typedef __attribute__((ext_vector_type(16))) _Float16     v16h;
typedef __attribute__((ext_vector_type(8)))  float        v8f;
typedef __attribute__((ext_vector_type(4)))  unsigned int v4u;
typedef __attribute__((ext_vector_type(8)))  int          v8i;
typedef __attribute__((ext_vector_type(4)))  int          v4i;

#define BB   2
#define LL   2048
#define MM   16
#define KTOP 32

#if defined(__has_builtin)
#if __has_builtin(__builtin_amdgcn_tensor_load_to_lds) && \
    __has_builtin(__builtin_amdgcn_s_wait_tensorcnt)
#define HAVE_TDM 1
#endif
#endif
#ifndef HAVE_TDM
#define HAVE_TDM 0
#endif

// pair list (atom-of-i, atom-of-j); atom ids: N=0, Ca=1, C=2, O=3, Cb=4
__constant__ int c_pairA[24] = {0,2,3,4,1,1,1,1,0,0,0,4,4,3,0,2,3,4,2,3,4,2,3,2};
__constant__ int c_pairB[24] = {0,2,3,4,0,2,3,4,2,3,4,2,3,2,1,1,1,1,0,0,0,4,4,3};

// ---- WMMA operand packing helpers (wave32, 16x16x32 f16) -------------------
// A-matrix 16x32: lanes 0-15 hold M=lane, K={0..7,16..23}; lanes 16-31 hold
// M=lane-16, K={8..15,24..31}; 16 halves per lane (2x b128 ds loads).
__device__ __forceinline__ int a_pack_idx(int m, int kk) {
  int lane, hlf;
  if      (kk <  8) { lane = m;      hlf = kk;      }
  else if (kk < 16) { lane = m + 16; hlf = kk - 8;  }
  else if (kk < 24) { lane = m;      hlf = kk - 8;  }
  else              { lane = m + 16; hlf = kk - 16; }
  return lane * 16 + hlf;
}

__device__ __forceinline__ float rbf_val(float D, int t) {
  float mu = 2.0f + 1.33333333333f * (float)t;   // linspace(2,22,16)
  float z  = (D - mu) * 0.8f;                    // sigma = 1.25
  return __expf(-z * z);
}

__device__ __forceinline__ v8f wmma_f16(v16h a, v16h b, v8f c) {
  return __builtin_amdgcn_wmma_f32_16x16x32_f16(false, a, false, b, (short)0, c,
                                                false, false);
}

// ---- Tensor Data Mover: 1D dword copy global -> LDS (TENSORcnt tracked) ----
#if HAVE_TDM
__device__ __forceinline__ void tdm_load_1d(const void* gsrc,
                                            unsigned int lds_byte_off,
                                            int ndwords) {
  unsigned long long ga = (unsigned long long)(size_t)gsrc;
  // D# group0: count=1 | lds_addr | global_addr[56:0] | type=2 (bits 127:126)
  v4u g0 = { 1u, lds_byte_off, (unsigned int)ga,
             (unsigned int)((ga >> 32) & 0x01FFFFFFu) | 0x80000000u };
  // D# group1: data_size=4B (code 2); tensor_dim0 = tile_dim0 = ndwords;
  // tensor_dim1 = tile_dim1 = 1; dim0 stride = ndwords.
  v8i g1 = { (int)(2u << 16),
             (int)(((unsigned)ndwords & 0xFFFFu) << 16),   // tensor_dim0 lo16
             (int)(((unsigned)ndwords >> 16) | (1u << 16)),// dim0 hi16 | dim1 lo
             (int)(((unsigned)ndwords & 0xFFFFu) << 16),   // tile_dim0
             1,                                            // tile_dim1
             ndwords,                                      // dim0 stride lo32
             0, 0 };
  v4i gz = {0, 0, 0, 0};
#if __clang_major__ >= 23
  v8i gz8 = {0, 0, 0, 0, 0, 0, 0, 0};
  __builtin_amdgcn_tensor_load_to_lds(g0, g1, gz, gz, gz8, 0);
#else
  __builtin_amdgcn_tensor_load_to_lds(g0, g1, gz, gz, 0);
#endif
}
#endif

// Issue one 8KB packed-B slice (2048 dwords) toward LDS. TDM: wave 0 only.
__device__ __forceinline__ void bslice_issue(const _Float16* src,
                                             _Float16* dstLDS, int tid) {
#if HAVE_TDM
  if (tid < 32) tdm_load_1d((const void*)src, (unsigned int)(size_t)dstLDS, 2048);
#else
  for (int x = tid; x < 2048; x += (int)blockDim.x)
    ((int*)dstLDS)[x] = ((const int*)src)[x];
#endif
}
__device__ __forceinline__ void bslice_wait_keep1() {   // oldest done, 1 in flight
#if HAVE_TDM
  __builtin_amdgcn_s_wait_tensorcnt(1);
#endif
}
__device__ __forceinline__ void bslice_wait_all() {
#if HAVE_TDM
  __builtin_amdgcn_s_wait_tensorcnt(0);
#endif
}

// ---- kernel 1: backbone atoms + virtual Cb ---------------------------------
__global__ __launch_bounds__(256) void atoms_kernel(const float* __restrict__ X,
                                                    float* __restrict__ atoms) {
  int r = blockIdx.x * 256 + threadIdx.x;
  if (r >= BB * LL) return;
  const float* x = X + (size_t)r * 12;
  float Nx=x[0],Ny=x[1],Nz=x[2], Cax=x[3],Cay=x[4],Caz=x[5];
  float Cx=x[6],Cy=x[7],Cz=x[8], Ox=x[9],Oy=x[10],Oz=x[11];
  float bx=Cax-Nx, by=Cay-Ny, bz=Caz-Nz;
  float cx=Cx-Cax, cy=Cy-Cay, cz=Cz-Caz;
  float ax=by*cz-bz*cy, ay=bz*cx-bx*cz, az=bx*cy-by*cx;
  float Cbx=-0.58273431f*ax+0.56802827f*bx-0.54067466f*cx+Cax;
  float Cby=-0.58273431f*ay+0.56802827f*by-0.54067466f*cy+Cay;
  float Cbz=-0.58273431f*az+0.56802827f*bz-0.54067466f*cz+Caz;
  float* o = atoms + (size_t)r * 15;
  o[0]=Nx;o[1]=Ny;o[2]=Nz; o[3]=Cax;o[4]=Cay;o[5]=Caz;
  o[6]=Cx;o[7]=Cy;o[8]=Cz; o[9]=Ox;o[10]=Oy;o[11]=Oz;
  o[12]=Cbx;o[13]=Cby;o[14]=Cbz;
}

// ---- kernel 2: pack weights into f16 WMMA B-tiles --------------------------
// B-matrix 32x16: lanes 0-15 -> N=lane, K=kt*32+{0..7,16..23};
//                 lanes 16-31 -> N=lane-16, K=kt*32+{8..15,24..31}.
__device__ __forceinline__ _Float16 pack_elem(const float* W, int Kdim, int x) {
  int tile = x >> 9, rem = x & 511, lane = rem >> 4, h = rem & 15;
  int kt = tile >> 3, nt = tile & 7;
  int n = nt * 16 + (lane & 15);
  int k = kt * 32 + ((lane >> 4) << 3) + (h < 8 ? h : h + 8);
  return (k < Kdim) ? (_Float16)W[n * Kdim + k] : (_Float16)0.0f;
}

__global__ __launch_bounds__(256) void pack_kernel(const float* __restrict__ We,
                                                   const float* __restrict__ Wd,
                                                   const float* __restrict__ Wy,
                                                   _Float16* __restrict__ pWe,
                                                   _Float16* __restrict__ pWd,
                                                   _Float16* __restrict__ pWy) {
  int gt = blockIdx.x * 256 + threadIdx.x;
  int nth = gridDim.x * 256;
  for (int x = gt; x < 13 * 8 * 512; x += nth) pWe[x] = pack_elem(We, 416, x);
  for (int x = gt; x <  5 * 8 * 512; x += nth) pWd[x] = pack_elem(Wd, 148, x);
  for (int x = gt; x <  1 * 8 * 512; x += nth) pWy[x] = pack_elem(Wy,  16, x);
}

// ---- kernel 3: masked pairwise distance + top-32 selection -----------------
__global__ __launch_bounds__(256) void topk_kernel(const float* __restrict__ atoms,
                                                   const float* __restrict__ mask,
                                                   int* __restrict__ eidx,
                                                   float* __restrict__ dnb,
                                                   float* __restrict__ outEidxF) {
  __shared__ float sD[LL];
  __shared__ float rv[256];
  __shared__ int   ri[256];
  int tid = threadIdx.x;
  int r = blockIdx.x;
  int bb = r / LL;
  const float* ai = atoms + (size_t)r * 15;
  float cax = ai[3], cay = ai[4], caz = ai[5];
  float mi = mask[r];
  float lmax = 0.0f;
  for (int j = tid; j < LL; j += 256) {
    int jn = j + 256 < LL ? j + 256 : j;
    __builtin_prefetch(atoms + ((size_t)bb * LL + jn) * 15, 0, 0);
    const float* aj = atoms + ((size_t)bb * LL + j) * 15;
    float dx = cax - aj[3], dy = cay - aj[4], dz = caz - aj[5];
    float m2 = mi * mask[bb * LL + j];
    float D = m2 * sqrtf(dx * dx + dy * dy + dz * dz + 1e-6f);
    sD[j] = D;
    lmax = fmaxf(lmax, D);
  }
  rv[tid] = lmax;
  __syncthreads();
  for (int s = 128; s > 0; s >>= 1) {
    if (tid < s) rv[tid] = fmaxf(rv[tid], rv[tid + s]);
    __syncthreads();
  }
  float Dmax = rv[0];
  __syncthreads();
  for (int j = tid; j < LL; j += 256) {
    float m2 = mi * mask[bb * LL + j];
    sD[j] += (1.0f - m2) * Dmax;
  }
  __syncthreads();
  for (int t = 0; t < KTOP; ++t) {
    float bv = 3.4e38f; int bi = LL;
    for (int j = tid; j < LL; j += 256)
      if (sD[j] < bv) { bv = sD[j]; bi = j; }
    rv[tid] = bv; ri[tid] = bi;
    __syncthreads();
    for (int s = 128; s > 0; s >>= 1) {
      if (tid < s) {
        if (rv[tid + s] < rv[tid] ||
            (rv[tid + s] == rv[tid] && ri[tid + s] < ri[tid])) {
          rv[tid] = rv[tid + s]; ri[tid] = ri[tid + s];
        }
      }
      __syncthreads();
    }
    if (tid == 0) {
      int wsel = ri[0];
      eidx[(size_t)r * KTOP + t] = wsel;
      dnb [(size_t)r * KTOP + t] = rv[0];
      outEidxF[(size_t)r * KTOP + t] = (float)wsel;
      sD[wsel] = 3.0e38f;
    }
    __syncthreads();
  }
}

// ---- kernel 4: fused edge features -> WMMA (416x128) -> layernorm ----------
// Block = 64 threads (2 waves) = 1 residue (32 edges). Packed W_edge streamed
// into LDS by the TDM in 8KB k-slices, double-buffered against the WMMAs.
__global__ __launch_bounds__(64) void edge_kernel(
    const float* __restrict__ atoms, const int* __restrict__ eidx,
    const float* __restrict__ dnb, const int* __restrict__ R_idx,
    const int* __restrict__ chains, const float* __restrict__ W_pos,
    const float* __restrict__ b_pos, const _Float16* __restrict__ pWe,
    const float* __restrict__ b_edge, const float* __restrict__ g_edge,
    const float* __restrict__ be_edge, float* __restrict__ outE) {
  __shared__ _Float16 sA[2 * 13 * 512];   // packed A tiles (26.6 KB)
  __shared__ _Float16 sB[2 * 4096];       // double-buffered B slices (16 KB)
  __shared__ float sAtomI[15];
  __shared__ float sAtomJ[32 * 15];
  __shared__ int   sJ[32];
  int tid = threadIdx.x;
  int r = blockIdx.x;                     // flattened b*L+l
  int bb = r / LL;
  bslice_issue(pWe, &sB[0], tid);         // slice 0 DMA overlaps feature gen
  if (tid < 15) sAtomI[tid] = atoms[(size_t)r * 15 + tid];
  if (tid < 32) sJ[tid] = eidx[(size_t)r * KTOP + tid];
  __syncthreads();
  for (int x = tid; x < 32 * 15; x += 64) {
    int e = x / 15, c = x % 15;
    sAtomJ[x] = atoms[((size_t)bb * LL + sJ[e]) * 15 + c];
  }
  __syncthreads();
  // features f[0..15] = E_pos (one-hot positional -> column gather of W_pos)
  for (int x = tid; x < 32 * 16; x += 64) {
    int e = x >> 4, c = x & 15;
    int j = bb * LL + sJ[e];
    int off = R_idx[r] - R_idx[j];
    int ch = (chains[r] == chains[j]) ? 1 : 0;
    int dc = off + 32; dc = dc < 0 ? 0 : (dc > 64 ? 64 : dc);
    int d = dc * ch + (1 - ch) * 65;
    float v = W_pos[c * 66 + d] + b_pos[c];
    sA[(e >> 4) * (13 * 512) + a_pack_idx(e & 15, c)] = (_Float16)v;
  }
  // features f[16..31] = rbf(D_neighbors)
  for (int x = tid; x < 32 * 16; x += 64) {
    int e = x >> 4, t = x & 15;
    float D = dnb[(size_t)r * KTOP + e];
    sA[(e >> 4) * (13 * 512) + a_pack_idx(e & 15, 16 + t)] = (_Float16)rbf_val(D, t);
  }
  // features f[32 + 16p + t] = rbf(dist(pairA_i, pairB_j))
  for (int x = tid; x < 32 * 24; x += 64) {
    int e = x % 32, p = x / 32;
    const float* Ai = &sAtomI[c_pairA[p] * 3];
    const float* Bj = &sAtomJ[e * 15 + c_pairB[p] * 3];
    float dx = Ai[0] - Bj[0], dy = Ai[1] - Bj[1], dz = Ai[2] - Bj[2];
    float D = sqrtf(dx * dx + dy * dy + dz * dz + 1e-6f);
    int fbase = 32 + p * 16;
    int abase = (e >> 4) * (13 * 512);
    for (int t = 0; t < 16; ++t) {
      int f = fbase + t;
      sA[abase + (f >> 5) * 512 + a_pack_idx(e & 15, f & 31)] =
          (_Float16)rbf_val(D, t);
    }
  }
  // WMMA: 13 k-slices x 8 n-tiles, TDM double-buffered B, f32 accumulate
  int wave = tid >> 5, lane = tid & 31;
  v8f acc[8] = {};
  for (int kt = 0; kt < 13; ++kt) {
    if (kt + 1 < 13) {
      bslice_issue(pWe + (kt + 1) * 4096, &sB[((kt + 1) & 1) * 4096], tid);
      bslice_wait_keep1();     // slice kt landed; slice kt+1 in flight
    } else {
      bslice_wait_all();
    }
    __syncthreads();
    v16h a = *(const v16h*)&sA[wave * (13 * 512) + kt * 512 + lane * 16];
    const _Float16* bslc = &sB[(kt & 1) * 4096];
#pragma unroll
    for (int nt = 0; nt < 8; ++nt) {
      v16h b = *(const v16h*)&bslc[nt * 512 + lane * 16];
      acc[nt] = wmma_f16(a, b, acc[nt]);
    }
    __syncthreads();           // done reading before this buffer is re-filled
  }
  // epilogue: +bias, layernorm over 128, scale/shift, store
  int hlf = lane >> 4, ln = lane & 15;
  float bvec[8], gvec[8], evec[8];
#pragma unroll
  for (int nt = 0; nt < 8; ++nt) {
    int n = nt * 16 + ln;
    bvec[nt] = b_edge[n]; gvec[nt] = g_edge[n]; evec[nt] = be_edge[n];
  }
#pragma unroll
  for (int rr = 0; rr < 8; ++rr) {
    int m = hlf * 8 + rr;
    float s = 0.0f, q = 0.0f;
#pragma unroll
    for (int nt = 0; nt < 8; ++nt) {
      float xv = acc[nt][rr] + bvec[nt];
      s += xv; q += xv * xv;
    }
    for (int o = 1; o < 16; o <<= 1) {
      s += __shfl_xor(s, o, 16);
      q += __shfl_xor(q, o, 16);
    }
    float mu = s * (1.0f / 128.0f);
    float var = q * (1.0f / 128.0f) - mu * mu;
    float inv = rsqrtf(var + 1e-5f);
    size_t ge = (size_t)r * KTOP + wave * 16 + m;
#pragma unroll
    for (int nt = 0; nt < 8; ++nt) {
      int n = nt * 16 + ln;
      outE[ge * 128 + n] = (acc[nt][rr] + bvec[nt] - mu) * inv * gvec[nt] + evec[nt];
    }
  }
}

// ---- kernel 5: fused node features -> WMMA (160x128) -> layernorm ----------
// Block = 256 threads (8 waves), one residue per wave; TDM-streamed W_down.
__global__ __launch_bounds__(256) void node_kernel(
    const float* __restrict__ atoms, const float* __restrict__ Y,
    const int* __restrict__ Y_t, const int* __restrict__ pt,
    const float* __restrict__ W_type, const float* __restrict__ b_type,
    const _Float16* __restrict__ pWd, const float* __restrict__ b_down,
    const float* __restrict__ g_node, const float* __restrict__ bn_node,
    float* __restrict__ outV) {
  __shared__ _Float16 sA[8 * 5 * 512];    // 40 KB
  __shared__ _Float16 sB[2 * 4096];       // 16 KB double buffer
  __shared__ float sFrame[8][12];         // e1,e2,e3,Ca per wave/residue
  int tid = threadIdx.x, wave = tid >> 5, lane = tid & 31;
  int r = blockIdx.x * 8 + wave;
  bslice_issue(pWd, &sB[0], tid);         // slice 0 DMA overlaps feature gen
  if (lane == 0) {
    const float* a = atoms + (size_t)r * 15;
    float v1x = a[0]-a[3], v1y = a[1]-a[4], v1z = a[2]-a[5];   // N - Ca
    float v2x = a[6]-a[3], v2y = a[7]-a[4], v2z = a[8]-a[5];   // C - Ca
    float n1 = sqrtf(v1x*v1x+v1y*v1y+v1z*v1z) + 1e-8f;
    float e1x=v1x/n1, e1y=v1y/n1, e1z=v1z/n1;
    float dp = e1x*v2x+e1y*v2y+e1z*v2z;
    float ux=v2x-e1x*dp, uy=v2y-e1y*dp, uz=v2z-e1z*dp;
    float n2 = sqrtf(ux*ux+uy*uy+uz*uz) + 1e-8f;
    float e2x=ux/n2, e2y=uy/n2, e2z=uz/n2;
    float e3x=e1y*e2z-e1z*e2y, e3y=e1z*e2x-e1x*e2z, e3z=e1x*e2y-e1y*e2x;
    sFrame[wave][0]=e1x; sFrame[wave][1]=e1y; sFrame[wave][2]=e1z;
    sFrame[wave][3]=e2x; sFrame[wave][4]=e2y; sFrame[wave][5]=e2z;
    sFrame[wave][6]=e3x; sFrame[wave][7]=e3y; sFrame[wave][8]=e3z;
    sFrame[wave][9]=a[3]; sFrame[wave][10]=a[4]; sFrame[wave][11]=a[5];
  }
  __syncthreads();
  int m = lane & 15, part = lane >> 4;
  _Float16* sAw = &sA[wave * (5 * 512)];
  float yx = Y[((size_t)r * 16 + m) * 3 + 0];
  float yy = Y[((size_t)r * 16 + m) * 3 + 1];
  float yz = Y[((size_t)r * 16 + m) * 3 + 2];
  if (part == 0) {
    // f[0..79]: rbf to N, Ca, C, O, Cb
    const float* a = atoms + (size_t)r * 15;
    for (int at = 0; at < 5; ++at) {
      float dx = a[at*3+0]-yx, dy = a[at*3+1]-yy, dz = a[at*3+2]-yz;
      float D = sqrtf(dx*dx+dy*dy+dz*dz + 1e-6f);
      for (int t = 0; t < 16; ++t) {
        int f = at * 16 + t;
        sAw[(f >> 5) * 512 + a_pack_idx(m, f & 31)] = (_Float16)rbf_val(D, t);
      }
    }
  } else {
    // f[80..143]: Y_t projection (one-hot -> 3-column gather of W_type)
    int t0 = Y_t[(size_t)r * 16 + m];
    int g1 = pt[120 + t0], g2 = pt[240 + t0];
    for (int c = 0; c < 64; ++c) {
      float v = W_type[c*147 + t0] + W_type[c*147 + 120 + g1] +
                W_type[c*147 + 139 + g2] + b_type[c];
      int f = 80 + c;
      sAw[(f >> 5) * 512 + a_pack_idx(m, f & 31)] = (_Float16)v;
    }
    // f[144..147]: angle features
    const float* F = sFrame[wave];
    float vx = yx - F[9], vy = yy - F[10], vz = yz - F[11];
    float l0 = F[0]*vx + F[1]*vy + F[2]*vz;
    float l1 = F[3]*vx + F[4]*vy + F[5]*vz;
    float l2 = F[6]*vx + F[7]*vy + F[8]*vz;
    float rxy  = sqrtf(l0*l0 + l1*l1 + 1e-8f);
    float rxyz = sqrtf(l0*l0 + l1*l1 + l2*l2) + 1e-8f;
    float fa[4] = {l0 / rxy, l1 / rxy, rxy / rxyz, l2 / rxyz};
    for (int c = 0; c < 4; ++c) {
      int f = 144 + c;
      sAw[(f >> 5) * 512 + a_pack_idx(m, f & 31)] = (_Float16)fa[c];
    }
    for (int f = 148; f < 160; ++f)   // K padding
      sAw[(f >> 5) * 512 + a_pack_idx(m, f & 31)] = (_Float16)0.0f;
  }
  v8f acc[8] = {};
  for (int kt = 0; kt < 5; ++kt) {
    if (kt + 1 < 5) {
      bslice_issue(pWd + (kt + 1) * 4096, &sB[((kt + 1) & 1) * 4096], tid);
      bslice_wait_keep1();
    } else {
      bslice_wait_all();
    }
    __syncthreads();
    v16h a = *(const v16h*)&sAw[kt * 512 + lane * 16];
    const _Float16* bslc = &sB[(kt & 1) * 4096];
#pragma unroll
    for (int nt = 0; nt < 8; ++nt) {
      v16h b = *(const v16h*)&bslc[nt * 512 + lane * 16];
      acc[nt] = wmma_f16(a, b, acc[nt]);
    }
    __syncthreads();
  }
  int hlf = lane >> 4, ln = lane & 15;
  float bvec[8], gvec[8], evec[8];
#pragma unroll
  for (int nt = 0; nt < 8; ++nt) {
    int n = nt * 16 + ln;
    bvec[nt] = b_down[n]; gvec[nt] = g_node[n]; evec[nt] = bn_node[n];
  }
#pragma unroll
  for (int rr = 0; rr < 8; ++rr) {
    int mm = hlf * 8 + rr;
    float s = 0.0f, q = 0.0f;
#pragma unroll
    for (int nt = 0; nt < 8; ++nt) {
      float xv = acc[nt][rr] + bvec[nt];
      s += xv; q += xv * xv;
    }
    for (int o = 1; o < 16; o <<= 1) {
      s += __shfl_xor(s, o, 16);
      q += __shfl_xor(q, o, 16);
    }
    float mu = s * (1.0f / 128.0f);
    float var = q * (1.0f / 128.0f) - mu * mu;
    float inv = rsqrtf(var + 1e-5f);
#pragma unroll
    for (int nt = 0; nt < 8; ++nt) {
      int n = nt * 16 + ln;
      outV[((size_t)r * 16 + mm) * 128 + n] =
          (acc[nt][rr] + bvec[nt] - mu) * inv * gvec[nt] + evec[nt];
    }
  }
}

// ---- kernel 6: fused Y-Y RBF -> WMMA (32x128) -> layernorm -----------------
// Block = 256 threads per residue: 256 (m,m') pairs = 16 A-tiles. W_ye (8KB)
// loaded once per block by the TDM, overlapped with the RBF phase.
__global__ __launch_bounds__(256) void yedge_kernel(
    const float* __restrict__ Y, const _Float16* __restrict__ pWy,
    const float* __restrict__ b_ye, const float* __restrict__ g_ye,
    const float* __restrict__ bn_ye, float* __restrict__ outYE) {
  __shared__ float sY[MM * 3];
  __shared__ _Float16 sA[16 * 512];   // 16 KB
  __shared__ _Float16 sB[4096];       // 8 KB packed W_ye
  int tid = threadIdx.x;
  bslice_issue(pWy, sB, tid);         // DMA overlaps RBF phase
  if (tid < MM * 3) sY[tid] = Y[(size_t)blockIdx.x * MM * 3 + tid];
  for (int x = tid; x < 16 * 512 / 2; x += 256) ((int*)sA)[x] = 0;  // K padding
  __syncthreads();
  {
    int m = tid >> 4, m2 = tid & 15;
    float dx = sY[m*3+0]-sY[m2*3+0], dy = sY[m*3+1]-sY[m2*3+1],
          dz = sY[m*3+2]-sY[m2*3+2];
    float D = sqrtf(dx*dx+dy*dy+dz*dz + 1e-6f);
    for (int t = 0; t < 16; ++t)
      sA[m * 512 + a_pack_idx(m2, t)] = (_Float16)rbf_val(D, t);
  }
  bslice_wait_all();
  __syncthreads();
  int wave = tid >> 5, lane = tid & 31;
  int hlf = lane >> 4, ln = lane & 15;
  float bvec[8], gvec[8], evec[8];
#pragma unroll
  for (int nt = 0; nt < 8; ++nt) {
    int n = nt * 16 + ln;
    bvec[nt] = b_ye[n]; gvec[nt] = g_ye[n]; evec[nt] = bn_ye[n];
  }
  for (int tt = wave; tt < 16; tt += 8) {
    v16h a = *(const v16h*)&sA[tt * 512 + lane * 16];
    v8f acc[8] = {};
#pragma unroll
    for (int nt = 0; nt < 8; ++nt) {
      v16h b = *(const v16h*)&sB[nt * 512 + lane * 16];
      acc[nt] = wmma_f16(a, b, acc[nt]);
    }
#pragma unroll
    for (int rr = 0; rr < 8; ++rr) {
      int m2 = hlf * 8 + rr;
      float s = 0.0f, q = 0.0f;
#pragma unroll
      for (int nt = 0; nt < 8; ++nt) {
        float xv = acc[nt][rr] + bvec[nt];
        s += xv; q += xv * xv;
      }
      for (int o = 1; o < 16; o <<= 1) {
        s += __shfl_xor(s, o, 16);
        q += __shfl_xor(q, o, 16);
      }
      float mu = s * (1.0f / 128.0f);
      float var = q * (1.0f / 128.0f) - mu * mu;
      float inv = rsqrtf(var + 1e-5f);
      size_t row = (size_t)blockIdx.x * 256 + tt * 16 + m2;
#pragma unroll
      for (int nt = 0; nt < 8; ++nt) {
        int n = nt * 16 + ln;
        outYE[row * 128 + n] =
            (acc[nt][rr] + bvec[nt] - mu) * inv * gvec[nt] + evec[nt];
      }
    }
  }
}

// ---- kernel 7: Y_nodes (one-hot -> 3-column gather) + layernorm; Y_m copy --
__global__ __launch_bounds__(256) void ynode_kernel(
    const int* __restrict__ Y_t, const int* __restrict__ pt,
    const float* __restrict__ W_yn, const float* __restrict__ b_yn,
    const float* __restrict__ g_yn, const float* __restrict__ bn_yn,
    const float* __restrict__ Y_m, float* __restrict__ outYN,
    float* __restrict__ outYM) {
  int tid = threadIdx.x, wave = tid >> 5, lane = tid & 31;
  size_t row = (size_t)blockIdx.x * 8 + wave;
  int t0 = Y_t[row];
  int g1 = pt[120 + t0], g2 = pt[240 + t0];
  float vals[4];
  float s = 0.0f, q = 0.0f;
#pragma unroll
  for (int k = 0; k < 4; ++k) {
    int c = lane + 32 * k;
    vals[k] = W_yn[c*147 + t0] + W_yn[c*147 + 120 + g1] +
              W_yn[c*147 + 139 + g2] + b_yn[c];
    s += vals[k]; q += vals[k] * vals[k];
  }
  for (int o = 1; o < 32; o <<= 1) {
    s += __shfl_xor(s, o, 32);
    q += __shfl_xor(q, o, 32);
  }
  float mu = s * (1.0f / 128.0f);
  float var = q * (1.0f / 128.0f) - mu * mu;
  float inv = rsqrtf(var + 1e-5f);
#pragma unroll
  for (int k = 0; k < 4; ++k) {
    int c = lane + 32 * k;
    outYN[row * 128 + c] = (vals[k] - mu) * inv * g_yn[c] + bn_yn[c];
  }
  if (lane == 0) outYM[row] = Y_m[row];
}

// ---- launcher --------------------------------------------------------------
extern "C" void kernel_launch(void* const* d_in, const int* in_sizes, int n_in,
                              void* d_out, int out_size, void* d_ws,
                              size_t ws_size, hipStream_t stream) {
  (void)in_sizes; (void)n_in; (void)out_size; (void)ws_size;
  const float* X        = (const float*)d_in[0];
  const float* Y        = (const float*)d_in[1];
  const float* Y_m      = (const float*)d_in[2];
  const float* mask     = (const float*)d_in[3];
  const float* W_pos    = (const float*)d_in[4];
  const float* b_pos    = (const float*)d_in[5];
  const float* W_edge   = (const float*)d_in[6];
  const float* b_edge   = (const float*)d_in[7];
  const float* g_edge   = (const float*)d_in[8];
  const float* be_edge  = (const float*)d_in[9];
  const float* W_down   = (const float*)d_in[10];
  const float* b_down   = (const float*)d_in[11];
  const float* g_node   = (const float*)d_in[12];
  const float* bn_node  = (const float*)d_in[13];
  const float* W_type   = (const float*)d_in[14];
  const float* b_type   = (const float*)d_in[15];
  const float* W_yn     = (const float*)d_in[16];
  const float* b_yn     = (const float*)d_in[17];
  const float* g_yn     = (const float*)d_in[18];
  const float* bn_yn    = (const float*)d_in[19];
  const float* W_ye     = (const float*)d_in[20];
  const float* b_ye     = (const float*)d_in[21];
  const float* g_ye     = (const float*)d_in[22];
  const float* bn_ye    = (const float*)d_in[23];
  const int*   Y_t      = (const int*)d_in[24];
  const int*   R_idx    = (const int*)d_in[25];
  const int*   chains   = (const int*)d_in[26];
  const int*   periodic = (const int*)d_in[27];

  // outputs: V, E, E_idx, Y_nodes, Y_edges, Y_m (flat, in return order)
  float* out = (float*)d_out;
  const size_t NV  = (size_t)BB * LL * MM * 128;
  const size_t NE  = (size_t)BB * LL * KTOP * 128;
  const size_t NEI = (size_t)BB * LL * KTOP;
  const size_t NYE = (size_t)BB * LL * MM * MM * 128;
  float* outV    = out;
  float* outE    = outV + NV;
  float* outEidx = outE + NE;
  float* outYN   = outEidx + NEI;
  float* outYE   = outYN + NV;
  float* outYM   = outYE + NYE;

  // workspace carve-out
  char* w = (char*)d_ws;
  size_t off = 0;
  auto take = [&](size_t bytes) -> char* {
    char* p = w + off;
    off = (off + bytes + 255) & ~(size_t)255;
    return p;
  };
  float*    atoms = (float*)take((size_t)BB * LL * 15 * 4);
  int*      eidx  = (int*)take((size_t)BB * LL * KTOP * 4);
  float*    dnb   = (float*)take((size_t)BB * LL * KTOP * 4);
  _Float16* pWe   = (_Float16*)take((size_t)13 * 8 * 512 * 2);
  _Float16* pWd   = (_Float16*)take((size_t)5 * 8 * 512 * 2);
  _Float16* pWy   = (_Float16*)take((size_t)1 * 8 * 512 * 2);

  atoms_kernel<<<(BB * LL + 255) / 256, 256, 0, stream>>>(X, atoms);
  pack_kernel<<<8, 256, 0, stream>>>(W_edge, W_down, W_ye, pWe, pWd, pWy);
  topk_kernel<<<BB * LL, 256, 0, stream>>>(atoms, mask, eidx, dnb, outEidx);
  edge_kernel<<<BB * LL, 64, 0, stream>>>(atoms, eidx, dnb, R_idx, chains,
                                          W_pos, b_pos, pWe, b_edge, g_edge,
                                          be_edge, outE);
  node_kernel<<<BB * LL / 8, 256, 0, stream>>>(atoms, Y, Y_t, periodic, W_type,
                                               b_type, pWd, b_down, g_node,
                                               bn_node, outV);
  yedge_kernel<<<BB * LL, 256, 0, stream>>>(Y, pWy, b_ye, g_ye, bn_ye, outYE);
  ynode_kernel<<<BB * LL * MM / 8, 256, 0, stream>>>(Y_t, periodic, W_yn, b_yn,
                                                     g_yn, bn_yn, Y_m, outYN,
                                                     outYM);
}
